// FcaBlock_11759620456608
// MI455X (gfx1250) — compile-verified, matrix operand order checked by reading, not run on hardware
//
#include <hip/hip_runtime.h>
#include <math.h>

// ---------------------------------------------------------------------------
// FcaBlock for MI455X (gfx1250): f16 WMMA GEMMs + flash attention + VALU glue
// ---------------------------------------------------------------------------

typedef __attribute__((ext_vector_type(16))) _Float16 v16h;
typedef __attribute__((ext_vector_type(8)))  float    v8f;

#define WMMA_F16(a, b, c) \
  __builtin_amdgcn_wmma_f32_16x16x32_f16(false, (a), false, (b), (short)0, (c), false, false)

// Problem constants
#define BATCH 16
#define DIM   384
#define HEADS 8
#define HD    48
#define HDP   64          // padded head dim for WMMA K
#define NTOK  784         // local tokens
#define NSUM  98          // summary tokens
#define NKEY  882         // 784 + 98
#define NKEYP 896         // padded to mult of 32
#define MROW  896         // padded per-batch row count for qkv input
#define MTOT  (BATCH * NTOK)   // 12544 (mult of 64)
#define MQKV  (BATCH * MROW)   // 14336 (mult of 64)
#define QSCALE 0.14433756729740643f
#define LNEPS 1e-5f

union FragU {
  v16h h;
  float4 f4[2];
};

// Load a 16-element A/B fragment for v_wmma_f32_16x16x32_f16.
// rowptr points at the start of this lane's row (M row for A, N row for B^T-
// stored-as-rows); data is K-contiguous. Layout per ISA 7.12.2:
//   element e of lane (g = lane/16) holds K = (e/8)*16 + g*8 + (e%8)
// -> two contiguous 16B chunks at k0+g*8 and k0+16+g*8.
__device__ __forceinline__ v16h load_frag(const _Float16* rowptr, int k0, int g) {
  FragU u;
  u.f4[0] = *reinterpret_cast<const float4*>(rowptr + k0 + g * 8);
  u.f4[1] = *reinterpret_cast<const float4*>(rowptr + k0 + 16 + g * 8);
  return u.h;
}

// 64x64 block GEMM core: C(64x64) += A(64xK) * B(64xK)^T, 4 waves of 32x32.
__device__ __forceinline__ void gemm_tile64(const _Float16* __restrict__ A, int lda,
                                            const _Float16* __restrict__ B, int ldb,
                                            int K, int mbase, int nbase,
                                            int ln, int g, v8f acc[2][2]) {
  const _Float16* a0p = A + (size_t)(mbase + ln) * lda;
  const _Float16* a1p = A + (size_t)(mbase + 16 + ln) * lda;
  const _Float16* b0p = B + (size_t)(nbase + ln) * ldb;
  const _Float16* b1p = B + (size_t)(nbase + 16 + ln) * ldb;
  for (int k0 = 0; k0 < K; k0 += 32) {
    v16h a0 = load_frag(a0p, k0, g);
    v16h a1 = load_frag(a1p, k0, g);
    v16h b0 = load_frag(b0p, k0, g);
    v16h b1 = load_frag(b1p, k0, g);
    acc[0][0] = WMMA_F16(a0, b0, acc[0][0]);
    acc[0][1] = WMMA_F16(a0, b1, acc[0][1]);
    acc[1][0] = WMMA_F16(a1, b0, acc[1][0]);
    acc[1][1] = WMMA_F16(a1, b1, acc[1][1]);
  }
}

__device__ __forceinline__ void gemm_prologue(int& ln, int& g, int& m0, int& n0) {
  int tid = threadIdx.x;
  int lane = tid & 31;
  int wave = tid >> 5;
  ln = lane & 15;
  g = lane >> 4;
  int wm = wave >> 1;
  int wn = wave & 1;
  m0 = blockIdx.y * 64 + wm * 32;
  n0 = blockIdx.x * 64 + wn * 32;
}

// --------------------------- utility kernels -------------------------------

__global__ void cvt_f16_kernel(const float* __restrict__ in, _Float16* __restrict__ out, int n) {
  int i = blockIdx.x * blockDim.x + threadIdx.x;
  if (i < n) out[i] = (_Float16)in[i];
}

__global__ void zero_f16_kernel(_Float16* __restrict__ p, size_t n) {
  size_t i = (size_t)blockIdx.x * blockDim.x + threadIdx.x;
  size_t st = (size_t)gridDim.x * blockDim.x;
  for (; i < n; i += st) p[i] = (_Float16)0.0f;
}

__device__ __forceinline__ float block_reduce_sum128(float v, float* red) {
  int tid = threadIdx.x;
  red[tid] = v;
  __syncthreads();
  for (int s = 64; s > 0; s >>= 1) {
    if (tid < s) red[tid] += red[tid + s];
    __syncthreads();
  }
  float r = red[0];
  __syncthreads();
  return r;
}

// LN of x (rows 0..783, norm1) and s*s_scale (rows 784..881, snorm) into a
// single padded (B*896, 384) f16 matrix; pad rows zeroed.
__global__ __launch_bounds__(128) void ln_in_kernel(
    const float* __restrict__ x, const float* __restrict__ s,
    const float* __restrict__ s_scale,
    const float* __restrict__ n1w, const float* __restrict__ n1b,
    const float* __restrict__ snw, const float* __restrict__ snb,
    _Float16* __restrict__ out) {
  __shared__ float red[128];
  __shared__ float vals[DIM];
  int row = blockIdx.x;
  int b = row / MROW, r = row % MROW;
  int tid = threadIdx.x;
  _Float16* orow = out + (size_t)row * DIM;
  if (r >= NKEY) {
    for (int i = tid; i < DIM; i += 128) orow[i] = (_Float16)0.0f;
    return;
  }
  const float *src, *w, *bi;
  float sc = 1.0f;
  if (r < NTOK) {
    src = x + ((size_t)b * NTOK + r) * DIM; w = n1w; bi = n1b;
  } else {
    src = s + ((size_t)b * NSUM + (r - NTOK)) * DIM; w = snw; bi = snb;
    sc = s_scale[r - NTOK];
  }
  float sum = 0.f;
  for (int i = tid; i < DIM; i += 128) { float v = src[i] * sc; vals[i] = v; sum += v; }
  float mean = block_reduce_sum128(sum, red) * (1.0f / DIM);
  float vs = 0.f;
  for (int i = tid; i < DIM; i += 128) { float d = vals[i] - mean; vs += d * d; }
  float inv = rsqrtf(block_reduce_sum128(vs, red) * (1.0f / DIM) + LNEPS);
  for (int i = tid; i < DIM; i += 128)
    orow[i] = (_Float16)((vals[i] - mean) * inv * w[i] + bi[i]);
}

// h = LN(LN(x2, norm2), mlp_norm) -> f16
__global__ __launch_bounds__(128) void double_ln_kernel(
    const float* __restrict__ x2,
    const float* __restrict__ n2w, const float* __restrict__ n2b,
    const float* __restrict__ mw, const float* __restrict__ mb,
    _Float16* __restrict__ out) {
  __shared__ float red[128];
  __shared__ float vals[DIM];
  int row = blockIdx.x;
  int tid = threadIdx.x;
  const float* src = x2 + (size_t)row * DIM;
  float sum = 0.f;
  for (int i = tid; i < DIM; i += 128) { float v = src[i]; vals[i] = v; sum += v; }
  float mean = block_reduce_sum128(sum, red) * (1.0f / DIM);
  float vs = 0.f;
  for (int i = tid; i < DIM; i += 128) { float d = vals[i] - mean; vs += d * d; }
  float inv = rsqrtf(block_reduce_sum128(vs, red) * (1.0f / DIM) + LNEPS);
  for (int i = tid; i < DIM; i += 128)
    vals[i] = (vals[i] - mean) * inv * n2w[i] + n2b[i];
  // second LN
  sum = 0.f;
  for (int i = tid; i < DIM; i += 128) sum += vals[i];
  mean = block_reduce_sum128(sum, red) * (1.0f / DIM);
  vs = 0.f;
  for (int i = tid; i < DIM; i += 128) { float d = vals[i] - mean; vs += d * d; }
  inv = rsqrtf(block_reduce_sum128(vs, red) * (1.0f / DIM) + LNEPS);
  _Float16* orow = out + (size_t)row * DIM;
  for (int i = tid; i < DIM; i += 128)
    orow[i] = (_Float16)((vals[i] - mean) * inv * mw[i] + mb[i]);
}

// --------------------------- GEMM kernels ----------------------------------

// QKV: (B*896, 384) @ (1152, 384)^T ; epilogue scatters into q (scaled),
// k, and v-transposed buffers padded to hd=64 / keys=896.
__global__ __launch_bounds__(128) void qkv_gemm_kernel(
    const _Float16* __restrict__ A, const _Float16* __restrict__ W,
    const float* __restrict__ bias,
    _Float16* __restrict__ qb, _Float16* __restrict__ kb, _Float16* __restrict__ vT) {
  int ln, g, m0, n0;
  gemm_prologue(ln, g, m0, n0);
  v8f z = {0.f, 0.f, 0.f, 0.f, 0.f, 0.f, 0.f, 0.f};
  v8f acc[2][2] = {{z, z}, {z, z}};
  gemm_tile64(A, DIM, W, DIM, DIM, m0, n0, ln, g, acc);
#pragma unroll
  for (int mt = 0; mt < 2; ++mt)
#pragma unroll
    for (int nt = 0; nt < 2; ++nt)
#pragma unroll
      for (int r = 0; r < 8; ++r) {
        int row = m0 + mt * 16 + g * 8 + r;
        int col = n0 + nt * 16 + ln;
        float val = acc[mt][nt][r] + bias[col];
        int b = row / MROW, rr = row % MROW;
        int which = col / DIM, rem = col % DIM;
        int h = rem / HD, d = rem % HD;
        size_t bh = (size_t)b * HEADS + h;
        if (which == 0) {
          if (rr < NTOK) qb[(bh * NTOK + rr) * HDP + d] = (_Float16)(val * QSCALE);
        } else if (which == 1) {
          if (rr < NKEY) kb[(bh * NKEYP + rr) * HDP + d] = (_Float16)val;
        } else {
          if (rr < NKEY) vT[(bh * HDP + d) * NKEYP + rr] = (_Float16)val;
        }
      }
}

// proj: (B*784, 384) @ (384, 384)^T ; x1 = x + gamma1 * (o + bias)
__global__ __launch_bounds__(128) void proj_gemm_kernel(
    const _Float16* __restrict__ A, const _Float16* __restrict__ W,
    const float* __restrict__ bias, const float* __restrict__ x_res,
    const float* __restrict__ gamma1, float* __restrict__ x1) {
  int ln, g, m0, n0;
  gemm_prologue(ln, g, m0, n0);
  v8f z = {0.f, 0.f, 0.f, 0.f, 0.f, 0.f, 0.f, 0.f};
  v8f acc[2][2] = {{z, z}, {z, z}};
  gemm_tile64(A, DIM, W, DIM, DIM, m0, n0, ln, g, acc);
#pragma unroll
  for (int mt = 0; mt < 2; ++mt)
#pragma unroll
    for (int nt = 0; nt < 2; ++nt)
#pragma unroll
      for (int r = 0; r < 8; ++r) {
        size_t row = m0 + mt * 16 + g * 8 + r;
        int col = n0 + nt * 16 + ln;
        float val = acc[mt][nt][r] + bias[col];
        x1[row * DIM + col] = x_res[row * DIM + col] + gamma1[col] * val;
      }
}

// fc1: (B*784, 384) @ (1536, 384)^T + bias, exact GELU, -> f16
__global__ __launch_bounds__(128) void fc1_gemm_kernel(
    const _Float16* __restrict__ A, const _Float16* __restrict__ W,
    const float* __restrict__ bias, _Float16* __restrict__ h1) {
  int ln, g, m0, n0;
  gemm_prologue(ln, g, m0, n0);
  v8f z = {0.f, 0.f, 0.f, 0.f, 0.f, 0.f, 0.f, 0.f};
  v8f acc[2][2] = {{z, z}, {z, z}};
  gemm_tile64(A, DIM, W, DIM, DIM, m0, n0, ln, g, acc);
#pragma unroll
  for (int mt = 0; mt < 2; ++mt)
#pragma unroll
    for (int nt = 0; nt < 2; ++nt)
#pragma unroll
      for (int r = 0; r < 8; ++r) {
        size_t row = m0 + mt * 16 + g * 8 + r;
        int col = n0 + nt * 16 + ln;
        float v = acc[mt][nt][r] + bias[col];
        float gel = 0.5f * v * (1.0f + erff(v * 0.70710678118654752f));
        h1[row * (4 * DIM) + col] = (_Float16)gel;
      }
}

// fc2: (B*784, 1536) @ (384, 1536)^T ; out = x2 + gamma2 * (h + bias)
__global__ __launch_bounds__(128) void fc2_gemm_kernel(
    const _Float16* __restrict__ A, const _Float16* __restrict__ W,
    const float* __restrict__ bias, const float* __restrict__ x2,
    const float* __restrict__ gamma2, float* __restrict__ out_x) {
  int ln, g, m0, n0;
  gemm_prologue(ln, g, m0, n0);
  v8f z = {0.f, 0.f, 0.f, 0.f, 0.f, 0.f, 0.f, 0.f};
  v8f acc[2][2] = {{z, z}, {z, z}};
  gemm_tile64(A, 4 * DIM, W, 4 * DIM, 4 * DIM, m0, n0, ln, g, acc);
#pragma unroll
  for (int mt = 0; mt < 2; ++mt)
#pragma unroll
    for (int nt = 0; nt < 2; ++nt)
#pragma unroll
      for (int r = 0; r < 8; ++r) {
        size_t row = m0 + mt * 16 + g * 8 + r;
        int col = n0 + nt * 16 + ln;
        float val = acc[mt][nt][r] + bias[col];
        out_x[row * DIM + col] = x2[row * DIM + col] + gamma2[col] * val;
      }
}

// --------------------------- attention kernel ------------------------------
// One wave = 16 query rows; flash-style online softmax over 28 chunks of
// 32 keys. QK^T and P@V both via v_wmma_f32_16x16x32_f16. P transposed
// C-layout -> A-layout through per-wave LDS scratch.
__global__ __launch_bounds__(128) void attn_kernel(
    const _Float16* __restrict__ q, const _Float16* __restrict__ k,
    const _Float16* __restrict__ vT,
    const float* __restrict__ attn_biases, const int* __restrict__ bias_idxs,
    const float* __restrict__ s_token_bias, _Float16* __restrict__ obuf) {
  __shared__ __align__(16) _Float16 plds[4][16][32];
  int tid = threadIdx.x;
  int lane = tid & 31, wave = tid >> 5;
  int ln = lane & 15, g = lane >> 4;
  int bh = blockIdx.x;
  int b = bh >> 3, h = bh & 7;
  int q0 = (blockIdx.y * 4 + wave) * 16;
  if (q0 >= NTOK) return;

  const _Float16* qbase = q + ((size_t)bh * NTOK + q0) * HDP;
  const _Float16* kbase = k + (size_t)bh * NKEYP * HDP;
  const _Float16* vbase = vT + (size_t)bh * HDP * NKEYP;

  v8f z = {0.f, 0.f, 0.f, 0.f, 0.f, 0.f, 0.f, 0.f};
  v8f oacc[4] = {z, z, z, z};
  float rmax[8], rsum[8];
#pragma unroll
  for (int r = 0; r < 8; ++r) { rmax[r] = -__builtin_inff(); rsum[r] = 0.f; }

  const _Float16* arow = qbase + (size_t)ln * HDP;

  for (int kt2 = 0; kt2 < NKEYP / 32; ++kt2) {
    v8f st[2];
#pragma unroll
    for (int t = 0; t < 2; ++t) {
      int kb = kt2 * 32 + t * 16;
      v8f sacc = z;
#pragma unroll
      for (int ks = 0; ks < 2; ++ks) {
        v16h a = load_frag(arow, ks * 32, g);
        v16h bb = load_frag(kbase + (size_t)(kb + ln) * HDP, ks * 32, g);
        sacc = WMMA_F16(a, bb, sacc);
      }
      int ng = kb + ln;
#pragma unroll
      for (int r = 0; r < 8; ++r) {
        int mg = q0 + g * 8 + r;
        if (ng >= NKEY) {
          sacc[r] = -__builtin_inff();
        } else if (ng >= NTOK) {
          sacc[r] += s_token_bias[h * NSUM + (ng - NTOK)];
        } else {
          int idx = bias_idxs[(size_t)mg * NTOK + ng];
          sacc[r] += attn_biases[h * NTOK + idx];
        }
      }
      st[t] = sacc;
    }

    // online softmax update over these 32 keys
    float nscale[8];
#pragma unroll
    for (int r = 0; r < 8; ++r) {
      float tm = fmaxf(st[0][r], st[1][r]);
      for (int o = 1; o < 16; o <<= 1) tm = fmaxf(tm, __shfl_xor(tm, o, 32));
      float mnew = fmaxf(rmax[r], tm);
      float sc = __expf(rmax[r] - mnew);
      float p0 = __expf(st[0][r] - mnew);
      float p1 = __expf(st[1][r] - mnew);
      float ps = p0 + p1;
      for (int o = 1; o < 16; o <<= 1) ps += __shfl_xor(ps, o, 32);
      rsum[r] = rsum[r] * sc + ps;
      rmax[r] = mnew;
      nscale[r] = sc;
      plds[wave][g * 8 + r][ln] = (_Float16)p0;
      plds[wave][g * 8 + r][16 + ln] = (_Float16)p1;
    }
#pragma unroll
    for (int nt = 0; nt < 4; ++nt)
#pragma unroll
      for (int r = 0; r < 8; ++r) oacc[nt][r] *= nscale[r];

    // ensure P writes are visible to this wave's fragment reads
    asm volatile("s_wait_dscnt 0" ::: "memory");

    v16h pA = load_frag(&plds[wave][ln][0], 0, g);
#pragma unroll
    for (int nt = 0; nt < 4; ++nt) {
      v16h vB = load_frag(vbase + (size_t)(nt * 16 + ln) * NKEYP, kt2 * 32, g);
      oacc[nt] = WMMA_F16(pA, vB, oacc[nt]);
    }
  }

#pragma unroll
  for (int r = 0; r < 8; ++r) rsum[r] = 1.0f / rsum[r];
#pragma unroll
  for (int nt = 0; nt < 4; ++nt) {
    int ng = nt * 16 + ln;
    if (ng >= HD) continue;
    int col = h * HD + ng;
#pragma unroll
    for (int r = 0; r < 8; ++r) {
      size_t row = (size_t)b * NTOK + q0 + g * 8 + r;
      obuf[row * DIM + col] = (_Float16)(oacc[nt][r] * rsum[r]);
    }
  }
}

// --------------------------- conv kernels ----------------------------------

// depthwise 7x7 stride 4 pad 3 over x1 image -> s_update (B,49,384)
__global__ void dwconv_kernel(const float* __restrict__ x1, const float* __restrict__ w,
                              float* __restrict__ out_s) {
  int idx = blockIdx.x * blockDim.x + threadIdx.x;
  if (idx >= BATCH * 49 * DIM) return;
  int c = idx % DIM;
  int t = idx / DIM;
  int p = t % 49, b = t / 49;
  int oh = p / 7, ow = p % 7;
  float acc = 0.f;
  for (int kh = 0; kh < 7; ++kh) {
    int ih = oh * 4 - 3 + kh;
    if (ih < 0 || ih >= 28) continue;
    for (int kw = 0; kw < 7; ++kw) {
      int iw = ow * 4 - 3 + kw;
      if (iw < 0 || iw >= 28) continue;
      acc += x1[((size_t)b * NTOK + ih * 28 + iw) * DIM + c] * w[c * 49 + kh * 7 + kw];
    }
  }
  out_s[idx] = acc;
}

// depthwise 3x3 stride 1 pad 1 + batchnorm -> x2
__global__ void lcbn_kernel(const float* __restrict__ x1, const float* __restrict__ w,
                            const float* __restrict__ bn_w, const float* __restrict__ bn_b,
                            const float* __restrict__ bn_mean, const float* __restrict__ bn_var,
                            float* __restrict__ x2) {
  int idx = blockIdx.x * blockDim.x + threadIdx.x;
  if (idx >= MTOT * DIM) return;
  int c = idx % DIM;
  int t = idx / DIM;
  int n = t % NTOK, b = t / NTOK;
  int oh = n / 28, ow = n % 28;
  float acc = 0.f;
  for (int kh = 0; kh < 3; ++kh) {
    int ih = oh - 1 + kh;
    if (ih < 0 || ih >= 28) continue;
    for (int kw = 0; kw < 3; ++kw) {
      int iw = ow - 1 + kw;
      if (iw < 0 || iw >= 28) continue;
      acc += x1[((size_t)b * NTOK + ih * 28 + iw) * DIM + c] * w[c * 9 + kh * 3 + kw];
    }
  }
  float inv = rsqrtf(bn_var[c] + LNEPS);
  x2[idx] = (acc - bn_mean[c]) * (inv * bn_w[c]) + bn_b[c];
}

// --------------------------- launcher --------------------------------------

extern "C" void kernel_launch(void* const* d_in, const int* in_sizes, int n_in,
                              void* d_out, int out_size, void* d_ws, size_t ws_size,
                              hipStream_t stream) {
  (void)in_sizes; (void)n_in; (void)out_size; (void)ws_size;
  const float* x        = (const float*)d_in[0];
  const float* s        = (const float*)d_in[1];
  const float* gamma1   = (const float*)d_in[2];
  const float* gamma2   = (const float*)d_in[3];
  const float* s_scale  = (const float*)d_in[4];
  const float* s_tok_b  = (const float*)d_in[5];
  const float* n1w      = (const float*)d_in[6];
  const float* n1b      = (const float*)d_in[7];
  const float* snw      = (const float*)d_in[8];
  const float* snb      = (const float*)d_in[9];
  const float* qkv_w    = (const float*)d_in[10];
  const float* qkv_b    = (const float*)d_in[11];
  const float* attn_b   = (const float*)d_in[12];
  const float* proj_w   = (const float*)d_in[13];
  const float* proj_b   = (const float*)d_in[14];
  const float* dw_w     = (const float*)d_in[15];
  const float* lc_w     = (const float*)d_in[16];
  const float* bn_w     = (const float*)d_in[17];
  const float* bn_b     = (const float*)d_in[18];
  const float* bn_mean  = (const float*)d_in[19];
  const float* bn_var   = (const float*)d_in[20];
  const float* n2w      = (const float*)d_in[21];
  const float* n2b      = (const float*)d_in[22];
  const float* mlw      = (const float*)d_in[23];
  const float* mlb      = (const float*)d_in[24];
  const float* fc1_w    = (const float*)d_in[25];
  const float* fc1_b    = (const float*)d_in[26];
  const float* fc2_w    = (const float*)d_in[27];
  const float* fc2_b    = (const float*)d_in[28];
  const int*   bias_idx = (const int*)d_in[29];

  float* out_x = (float*)d_out;
  float* out_s = out_x + (size_t)MTOT * DIM;

  char* base = (char*)d_ws;
  size_t off = 0;
  auto take = [&](size_t bytes) -> char* {
    char* p = base + off;
    off = (off + bytes + 255) & ~(size_t)255;
    return p;
  };

  const size_t XSN_N  = (size_t)MQKV * DIM;
  const size_t Q_N    = (size_t)BATCH * HEADS * NTOK * HDP;
  const size_t K_N    = (size_t)BATCH * HEADS * NKEYP * HDP;
  const size_t VT_N   = K_N;
  const size_t OB_N   = (size_t)MTOT * DIM;
  const size_t H16_N  = (size_t)MTOT * DIM;
  const size_t H1_N   = (size_t)MTOT * 4 * DIM;
  const int WQKV_N = 3 * DIM * DIM;
  const int WPROJ_N = DIM * DIM;
  const int WFC1_N = 4 * DIM * DIM;
  const int WFC2_N = DIM * 4 * DIM;

  _Float16* xs_n  = (_Float16*)take(XSN_N * 2);
  _Float16* qb    = (_Float16*)take(Q_N * 2);
  _Float16* kb    = (_Float16*)take(K_N * 2);
  _Float16* vT    = (_Float16*)take(VT_N * 2);
  _Float16* obuf  = (_Float16*)take(OB_N * 2);
  _Float16* h16   = (_Float16*)take(H16_N * 2);
  _Float16* h1    = (_Float16*)take(H1_N * 2);
  _Float16* wqkv  = (_Float16*)take((size_t)WQKV_N * 2);
  _Float16* wproj = (_Float16*)take((size_t)WPROJ_N * 2);
  _Float16* wfc1  = (_Float16*)take((size_t)WFC1_N * 2);
  _Float16* wfc2  = (_Float16*)take((size_t)WFC2_N * 2);
  float*    x1    = (float*)take((size_t)MTOT * DIM * 4);
  float*    x2    = (float*)take((size_t)MTOT * DIM * 4);

  // 1) weight conversion f32 -> f16
  cvt_f16_kernel<<<(WQKV_N + 255) / 256, 256, 0, stream>>>(qkv_w, wqkv, WQKV_N);
  cvt_f16_kernel<<<(WPROJ_N + 255) / 256, 256, 0, stream>>>(proj_w, wproj, WPROJ_N);
  cvt_f16_kernel<<<(WFC1_N + 255) / 256, 256, 0, stream>>>(fc1_w, wfc1, WFC1_N);
  cvt_f16_kernel<<<(WFC2_N + 255) / 256, 256, 0, stream>>>(fc2_w, wfc2, WFC2_N);

  // 2) zero the padded q/k/vT region (contiguous)
  zero_f16_kernel<<<2048, 256, 0, stream>>>(qb, Q_N + K_N + VT_N);

  // 3) LN(x) / LN(s * s_scale) -> padded f16 matrix
  ln_in_kernel<<<MQKV, 128, 0, stream>>>(x, s, s_scale, n1w, n1b, snw, snb, xs_n);

  // 4) QKV GEMM (M=14336, N=1152, K=384)
  qkv_gemm_kernel<<<dim3(1152 / 64, MQKV / 64), 128, 0, stream>>>(
      xs_n, wqkv, qkv_b, qb, kb, vT);

  // 5) flash attention
  attn_kernel<<<dim3(BATCH * HEADS, (NTOK / 16 + 3) / 4), 128, 0, stream>>>(
      qb, kb, vT, attn_b, bias_idx, s_tok_b, obuf);

  // 6) proj + gamma1 residual (M=12544, N=384, K=384)
  proj_gemm_kernel<<<dim3(DIM / 64, MTOT / 64), 128, 0, stream>>>(
      obuf, wproj, proj_b, x, gamma1, x1);

  // 7) depthwise 7x7 s4 -> s_update output
  dwconv_kernel<<<(BATCH * 49 * DIM + 255) / 256, 256, 0, stream>>>(x1, dw_w, out_s);

  // 8) depthwise 3x3 + BN -> x2
  lcbn_kernel<<<((size_t)MTOT * DIM + 255) / 256, 256, 0, stream>>>(
      x1, lc_w, bn_w, bn_b, bn_mean, bn_var, x2);

  // 9) double LN -> f16
  double_ln_kernel<<<MTOT, 128, 0, stream>>>(x2, n2w, n2b, mlw, mlb, h16);

  // 10) fc1 + exact GELU (M=12544, N=1536, K=384)
  fc1_gemm_kernel<<<dim3(4 * DIM / 64, MTOT / 64), 128, 0, stream>>>(h16, wfc1, fc1_b, h1);

  // 11) fc2 + gamma2 residual -> final x output (M=12544, N=384, K=1536)
  fc2_gemm_kernel<<<dim3(DIM / 64, MTOT / 64), 128, 0, stream>>>(
      h1, wfc2, fc2_b, x2, gamma2, out_x);
}
